// DecayModel_67705864454819
// MI455X (gfx1250) — compile-verified, matrix operand order checked by reading, not run on hardware
//
#include <hip/hip_runtime.h>

// ============================================================================
// DecayModel forward for MI455X (gfx1250): bf16 WMMA GEMM core for all
// conv/projection/MLP work, f32 for statistics, sparsity-exploiting
// time-domain convolutions instead of FFTs (am is 16-sparse per batch).
// Round 2: dword-vectorized LDS staging everywhere; dilated conv uses a
// halo-extended shared segment (one activation read serves all 3 taps)
// with a 128-wide time tile (8 WMMAs per wave per K-chunk).
// ============================================================================

typedef __attribute__((ext_vector_type(16))) __bf16 v16bf;
typedef __attribute__((ext_vector_type(8)))  float  v8f;
typedef unsigned short u16;
typedef unsigned int   u32;

union FragAB { v16bf v; u16 u[16]; };
union FragC  { v8f  v; float f[8]; };

#define T_N   32768
#define BATCH 8
#define CH    128
#define SA    36    // LDS A-tile row stride (u16), padded vs bank conflicts
#define SB    72    // LDS B-tile row stride (u16)
#define SEG   296   // LDS halo segment row stride (u16): 128 + 2*82 = 292 used

__device__ __forceinline__ u16 f2bfu(float f){
  u32 u = __float_as_uint(f);
  return (u16)((u + 0x7FFFu + ((u >> 16) & 1u)) >> 16);   // RNE
}
__device__ __forceinline__ float bfu2f(u16 s){ return __uint_as_float(((u32)s) << 16); }
__device__ __forceinline__ float lrelu_f(float x){ return x > 0.f ? x : 0.01f * x; }

// WMMA bf16 fragment index helper (ISA 7.12.2, 16-bit A/B 16x32 layout).
__device__ __forceinline__ int frag_k(int lane, int e){
  int kb = (lane & 16) ? 8 : 0;
  return kb + e + ((e & 8) ? 8 : 0);   // e<8 -> kb+e ; e>=8 -> kb+e+8
}
__device__ __forceinline__ v8f wmma_bf16(const FragAB& a, const FragAB& b, v8f c){
  return __builtin_amdgcn_wmma_f32_16x16x32_bf16(false, a.v, false, b.v,
                                                 (short)0, c, false, false);
}

// ---------------------------------------------------------------------------
// Packing kernels: f32 weights -> bf16 in WMMA-friendly layouts
// ---------------------------------------------------------------------------
__global__ void pack_bf_kernel(const float* __restrict__ s, u16* __restrict__ d, int n){
  int i = blockIdx.x * 256 + threadIdx.x;
  if (i < n) d[i] = f2bfu(s[i]);
}
__global__ void pack_pad_kernel(const float* __restrict__ s, u16* __restrict__ d,
                                int O, int Is, int Id){
  int i = blockIdx.x * 256 + threadIdx.x;
  if (i < O * Id){
    int o = i / Id, c = i - o * Id;
    d[i] = (c < Is) ? f2bfu(s[o * Is + c]) : (u16)0;
  }
}
// dst[k*O + n] = src[n*I + k]   (w[out,in] -> B[in][out])
__global__ void pack_tr_kernel(const float* __restrict__ s, u16* __restrict__ d,
                               int O, int I){
  int i = blockIdx.x * 256 + threadIdx.x;
  if (i < I * O){
    int k = i / O, n = i - k * O;
    d[i] = f2bfu(s[n * I + k]);
  }
}
// dil_w [6][128][128][3] -> A [6][128][j*128+c]
__global__ void pack_conv_kernel(const float* __restrict__ s, u16* __restrict__ d){
  int i = blockIdx.x * 256 + threadIdx.x;
  if (i < 6 * 128 * 384){
    int l = i / (128 * 384);
    int rem = i - l * 128 * 384;
    int o = rem / 384;
    int kk = rem - o * 384;
    int j = kk >> 7, c = kk & 127;
    d[i] = f2bfu(s[((l * 128 + o) * 128 + c) * 3 + j]);
  }
}

// ---------------------------------------------------------------------------
// Positional encoding table [33][T] bf16 (row0=p, 1..16 sin, 17..32 cos)
// ---------------------------------------------------------------------------
__global__ __launch_bounds__(256) void pe_kernel(u16* __restrict__ pe){
  int t = blockIdx.x * 256 + threadIdx.x;
  float p = -1.f + 2.f * (float)t / (float)(T_N - 1);
  pe[t] = f2bfu(p);
  const float PI = 3.14159265358979f;
  float f = 1.f;
  for (int q = 0; q < 16; ++q){
    float a = p * f * PI;
    pe[(1 + q)  * T_N + t] = f2bfu(sinf(a));
    pe[(17 + q) * T_N + t] = f2bfu(cosf(a));
    f *= 2.f;
  }
}

// ---------------------------------------------------------------------------
// Filterbank conv as WMMA GEMM: A = W[128x512], B Toeplitz window of x.
// out[b][c][t] = sum_k x[b, t+k-255] * w[c,k]   (jax SAME, K=512)
// ---------------------------------------------------------------------------
__global__ __launch_bounds__(256) void fb_conv_kernel(const float* __restrict__ x,
    const u16* __restrict__ A, u16* __restrict__ out){
  __shared__ u16 As[128 * SA];
  __shared__ u16 xseg[576];
  const int b  = blockIdx.z;
  const int t0 = blockIdx.x * 64;
  const int tid = threadIdx.x;
  const int wv = tid >> 5, lane = tid & 31;
  for (int i = tid; i < 575; i += 256){
    int g = t0 + i - 255;
    xseg[i] = f2bfu((g >= 0 && g < T_N) ? x[b * T_N + g] : 0.f);
  }
  FragC acc[4];
  #pragma unroll
  for (int j = 0; j < 4; ++j)
    #pragma unroll
    for (int e = 0; e < 8; ++e) acc[j].f[e] = 0.f;

  for (int k0 = 0; k0 < 512; k0 += 32){
    __syncthreads();
    for (int l = tid; l < 128 * 16; l += 256){      // u32 staging: 2 u16/load
      int r = l >> 4, cu = l & 15;
      ((u32*)(As + r * SA))[cu] = ((const u32*)(A + r * 512 + k0))[cu];
    }
    __syncthreads();
    FragAB fa;
    int m = wv * 16 + (lane & 15);
    #pragma unroll
    for (int e = 0; e < 16; ++e) fa.u[e] = As[m * SA + frag_k(lane, e)];
    #pragma unroll
    for (int j = 0; j < 4; ++j){
      FragAB fb;
      int n = j * 16 + (lane & 15);
      #pragma unroll
      for (int e = 0; e < 16; ++e) fb.u[e] = xseg[n + k0 + frag_k(lane, e)];
      acc[j].v = wmma_bf16(fa, fb, acc[j].v);
    }
  }
  const int mhalf = (lane >> 4) << 3;
  #pragma unroll
  for (int j = 0; j < 4; ++j)
    #pragma unroll
    for (int g = 0; g < 8; ++g){
      int m = wv * 16 + mhalf + g;
      int n = t0 + j * 16 + (lane & 15);
      out[(b * CH + m) * T_N + n] = f2bfu(acc[j].f[g]);
    }
}

// ---------------------------------------------------------------------------
// Reduce projection: out = Wr[128x192] * concat(pe(33), h(128), zeros(31))
// ---------------------------------------------------------------------------
__global__ __launch_bounds__(256) void reduce_kernel(const u16* __restrict__ pe,
    const u16* __restrict__ hin, const u16* __restrict__ A,
    const float* __restrict__ bias_m, u16* __restrict__ hout){
  __shared__ u16 As[128 * SA];
  __shared__ u16 Bs[32 * SB];
  const int b  = blockIdx.z;
  const int t0 = blockIdx.x * 64;
  const int tid = threadIdx.x;
  const int wv = tid >> 5, lane = tid & 31;
  FragC acc[4];
  #pragma unroll
  for (int j = 0; j < 4; ++j)
    #pragma unroll
    for (int e = 0; e < 8; ++e) acc[j].f[e] = 0.f;

  for (int k0 = 0; k0 < 192; k0 += 32){
    __syncthreads();
    for (int l = tid; l < 128 * 16; l += 256){
      int r = l >> 4, cu = l & 15;
      ((u32*)(As + r * SA))[cu] = ((const u32*)(A + r * 192 + k0))[cu];
    }
    for (int l = tid; l < 32 * 32; l += 256){       // u32 staging
      int r = l >> 5, cu = l & 31;
      int k = k0 + r;
      u32 val;
      if (k < 33)       val = ((const u32*)(pe + k * T_N + t0))[cu];
      else if (k < 161) val = ((const u32*)(hin + (b * CH + (k - 33)) * T_N + t0))[cu];
      else              val = 0u;
      ((u32*)(Bs + r * SB))[cu] = val;
    }
    __syncthreads();
    FragAB fa;
    int m = wv * 16 + (lane & 15);
    #pragma unroll
    for (int e = 0; e < 16; ++e) fa.u[e] = As[m * SA + frag_k(lane, e)];
    #pragma unroll
    for (int j = 0; j < 4; ++j){
      FragAB fb;
      #pragma unroll
      for (int e = 0; e < 16; ++e)
        fb.u[e] = Bs[frag_k(lane, e) * SB + j * 16 + (lane & 15)];
      acc[j].v = wmma_bf16(fa, fb, acc[j].v);
    }
  }
  const int mhalf = (lane >> 4) << 3;
  #pragma unroll
  for (int j = 0; j < 4; ++j)
    #pragma unroll
    for (int g = 0; g < 8; ++g){
      int m = wv * 16 + mhalf + g;
      int n = t0 + j * 16 + (lane & 15);
      hout[(b * CH + m) * T_N + n] = f2bfu(acc[j].f[g] + bias_m[m]);
    }
}

// ---------------------------------------------------------------------------
// Dilated residual conv layer, 128-wide time tile.
// out = lrelu(conv3(h,dil) + bias) + h
// One halo-extended segment per 32-channel block serves all 3 taps:
// seg[r][i] = h[b, cb*32+r, t0 - 82 + i],  i in [0,292)  (u32-aligned loads)
// ---------------------------------------------------------------------------
__global__ __launch_bounds__(256) void dconv_kernel(const u16* __restrict__ hin,
    const u16* __restrict__ A, const float* __restrict__ bias_m,
    u16* __restrict__ hout, int dil){
  __shared__ u16 As[128 * SA];
  __shared__ u16 seg[32 * SEG];
  const int b  = blockIdx.z;
  const int t0 = blockIdx.x * 128;
  const int tid = threadIdx.x;
  const int wv = tid >> 5, lane = tid & 31;
  const int halo = 82;                               // even, >= max dil (81)
  __builtin_prefetch(hin + ((size_t)(b * CH + (tid & 127)) * T_N + t0), 0, 1);

  FragC acc[8];
  #pragma unroll
  for (int j = 0; j < 8; ++j)
    #pragma unroll
    for (int e = 0; e < 8; ++e) acc[j].f[e] = 0.f;

  for (int cb = 0; cb < 4; ++cb){
    __syncthreads();
    // stage halo segment: 32 rows x 146 u32 (292 u16), dword-aligned
    for (int l = tid; l < 32 * 146; l += 256){
      int r = l / 146, cu = l - r * 146;
      int tb = t0 - halo + cu * 2;
      int rowb = (b * CH + cb * 32 + r) * T_N;
      u32 val;
      if (tb >= 0 && tb + 1 < T_N){
        val = *(const u32*)(hin + rowb + tb);
      } else {
        u16 lo16 = (tb     >= 0 && tb     < T_N) ? hin[rowb + tb]     : (u16)0;
        u16 hi16 = (tb + 1 >= 0 && tb + 1 < T_N) ? hin[rowb + tb + 1] : (u16)0;
        val = (u32)lo16 | ((u32)hi16 << 16);
      }
      ((u32*)(seg + r * SEG))[cu] = val;
    }
    for (int j = 0; j < 3; ++j){
      __syncthreads();
      const int k0 = j * 128 + cb * 32;
      for (int l = tid; l < 128 * 16; l += 256){
        int r = l >> 4, cu = l & 15;
        ((u32*)(As + r * SA))[cu] = ((const u32*)(A + r * 384 + k0))[cu];
      }
      __syncthreads();
      FragAB fa;
      int m = wv * 16 + (lane & 15);
      #pragma unroll
      for (int e = 0; e < 16; ++e) fa.u[e] = As[m * SA + frag_k(lane, e)];
      const int soff = halo + (j - 1) * dil;         // in [1, 163]
      #pragma unroll
      for (int jn = 0; jn < 8; ++jn){
        FragAB fb;
        int n = jn * 16 + (lane & 15);
        #pragma unroll
        for (int e = 0; e < 16; ++e)
          fb.u[e] = seg[frag_k(lane, e) * SEG + n + soff];
        acc[jn].v = wmma_bf16(fa, fb, acc[jn].v);
      }
    }
  }
  const int mhalf = (lane >> 4) << 3;
  #pragma unroll
  for (int jn = 0; jn < 8; ++jn)
    #pragma unroll
    for (int g = 0; g < 8; ++g){
      int m = wv * 16 + mhalf + g;
      int n = t0 + jn * 16 + (lane & 15);
      float v = lrelu_f(acc[jn].f[g] + bias_m[m]) + bfu2f(hin[(b * CH + m) * T_N + n]);
      hout[(b * CH + m) * T_N + n] = f2bfu(v);
    }
}

// ---------------------------------------------------------------------------
// Generic small GEMM (MLP stacks): C[M,N] = act(A[M,K] * B[K,N] + bias_n)
// M <= 128 (rows beyond M staged as zero), K % 32 == 0.
// ---------------------------------------------------------------------------
__global__ __launch_bounds__(256) void gemm_stack_kernel(const u16* __restrict__ A,
    const u16* __restrict__ B, const float* __restrict__ bias_n,
    int M, int N, int K, int act,
    float* __restrict__ outF, u16* __restrict__ outB){
  __shared__ u16 As[128 * SA];
  __shared__ u16 Bs[32 * SB];
  const int n0 = blockIdx.x * 64;
  const int tid = threadIdx.x;
  const int wv = tid >> 5, lane = tid & 31;
  FragC acc[4];
  #pragma unroll
  for (int j = 0; j < 4; ++j)
    #pragma unroll
    for (int e = 0; e < 8; ++e) acc[j].f[e] = 0.f;

  for (int k0 = 0; k0 < K; k0 += 32){
    __syncthreads();
    for (int l = tid; l < 128 * 16; l += 256){
      int r = l >> 4, cu = l & 15;
      ((u32*)(As + r * SA))[cu] =
          (r < M) ? ((const u32*)(A + r * K + k0))[cu] : 0u;
    }
    if ((N & 63) == 0){
      for (int l = tid; l < 32 * 32; l += 256){
        int r = l >> 5, cu = l & 31;
        ((u32*)(Bs + r * SB))[cu] = ((const u32*)(B + (k0 + r) * N + n0))[cu];
      }
    } else {
      for (int l = tid; l < 32 * 64; l += 256){
        int r = l >> 6, c = l & 63;
        Bs[r * SB + c] = (n0 + c < N) ? B[(k0 + r) * N + n0 + c] : (u16)0;
      }
    }
    __syncthreads();
    FragAB fa;
    int m = wv * 16 + (lane & 15);
    #pragma unroll
    for (int e = 0; e < 16; ++e) fa.u[e] = As[m * SA + frag_k(lane, e)];
    #pragma unroll
    for (int j = 0; j < 4; ++j){
      FragAB fb;
      #pragma unroll
      for (int e = 0; e < 16; ++e)
        fb.u[e] = Bs[frag_k(lane, e) * SB + j * 16 + (lane & 15)];
      acc[j].v = wmma_bf16(fa, fb, acc[j].v);
    }
  }
  const int mhalf = (lane >> 4) << 3;
  #pragma unroll
  for (int j = 0; j < 4; ++j)
    #pragma unroll
    for (int g = 0; g < 8; ++g){
      int m = wv * 16 + mhalf + g;
      int n = n0 + j * 16 + (lane & 15);
      if (m < M && n < N){
        float v = acc[j].f[g] + (bias_n ? bias_n[n] : 0.f);
        if (act) v = lrelu_f(v);
        if (outF) outF[m * N + n] = v;
        if (outB) outB[m * N + n] = f2bfu(v);
      }
    }
}

// ---------------------------------------------------------------------------
// Per-row (b,c) sum / sumsq for ExampleNorm + context mean (u32 pair reads)
// ---------------------------------------------------------------------------
__global__ __launch_bounds__(256) void rowstat_kernel(const u16* __restrict__ h,
    float* __restrict__ rowsum, float* __restrict__ rowsq){
  __shared__ float s1[256], s2[256];
  int row = blockIdx.x;
  const u32* src = (const u32*)(h + (size_t)row * T_N);
  float a = 0.f, q = 0.f;
  for (int i = threadIdx.x; i < T_N / 2; i += 256){
    u32 w = src[i];
    float v0 = bfu2f((u16)(w & 0xffffu));
    float v1 = bfu2f((u16)(w >> 16));
    a += v0 + v1; q += v0 * v0 + v1 * v1;
  }
  s1[threadIdx.x] = a; s2[threadIdx.x] = q;
  __syncthreads();
  for (int s = 128; s > 0; s >>= 1){
    if (threadIdx.x < s){ s1[threadIdx.x] += s1[threadIdx.x + s];
                          s2[threadIdx.x] += s2[threadIdx.x + s]; }
    __syncthreads();
  }
  if (threadIdx.x == 0){ rowsum[row] = s1[0]; rowsq[row] = s2[0]; }
}

__global__ __launch_bounds__(128) void batchstat_kernel(const float* __restrict__ rowsum,
    const float* __restrict__ rowsq, float* __restrict__ inv_std,
    u16* __restrict__ ctx_in){
  int b = blockIdx.x;     // grid 16: rows 8..15 of ctx_in zero-padded
  int c = threadIdx.x;
  if (b >= 8){ ctx_in[b * 128 + c] = 0; return; }
  __shared__ float s1[128], s2[128];
  s1[c] = rowsum[b * 128 + c]; s2[c] = rowsq[b * 128 + c];
  __syncthreads();
  for (int s = 64; s > 0; s >>= 1){
    if (c < s){ s1[c] += s1[c + s]; s2[c] += s2[c + s]; }
    __syncthreads();
  }
  float cnt = 128.f * (float)T_N;
  float mean = s1[0] / cnt;
  float var  = s2[0] / cnt - mean * mean;
  float inv  = 1.f / (sqrtf(fmaxf(var, 0.f)) + 1e-8f);
  if (c == 0) inv_std[b] = inv;
  ctx_in[b * 128 + c] = f2bfu(rowsum[b * 128 + c] / (float)T_N * inv);
}

__global__ __launch_bounds__(256) void scale_kernel(const u16* __restrict__ in,
    u16* __restrict__ out, const float* __restrict__ inv_std){
  size_t i = (size_t)blockIdx.x * 256 + threadIdx.x;   // over u32 pairs
  int b = (int)(i >> 21);                 // (2i) >> 22
  float s = inv_std[b];
  u32 w = ((const u32*)in)[i];
  float v0 = bfu2f((u16)(w & 0xffffu)) * s;
  float v1 = bfu2f((u16)(w >> 16)) * s;
  ((u32*)out)[i] = (u32)f2bfu(v0) | ((u32)f2bfu(v1) << 16);
}

// ---------------------------------------------------------------------------
// Attention (1) + atom-map (16) projections from normalized h
// ---------------------------------------------------------------------------
__global__ __launch_bounds__(128) void attn_am_kernel(const u16* __restrict__ hN,
    const float* __restrict__ attn_w, const float* __restrict__ attn_b,
    const float* __restrict__ atom_w, const float* __restrict__ atom_b,
    float* __restrict__ attn, float* __restrict__ am){
  __shared__ float sw[128 + 16 * 128];
  int tid = threadIdx.x;
  sw[tid] = attn_w[tid];
  for (int i = tid; i < 2048; i += 128) sw[128 + i] = atom_w[i];
  __syncthreads();
  int b = blockIdx.y;
  int t = blockIdx.x * 128 + tid;
  float a0 = 0.f;
  float aj[16];
  #pragma unroll
  for (int j = 0; j < 16; ++j) aj[j] = 0.f;
  const u16* hp = hN + (size_t)b * CH * T_N + t;
  for (int c = 0; c < 128; ++c){
    float hv = bfu2f(hp[c * T_N]);
    a0 += sw[c] * hv;
    #pragma unroll
    for (int j = 0; j < 16; ++j) aj[j] += sw[128 + j * 128 + c] * hv;
  }
  attn[b * T_N + t] = a0 + attn_b[0];
  #pragma unroll
  for (int j = 0; j < 16; ++j) am[(b * 16 + j) * T_N + t] = aj[j] + atom_b[j];
}

// ---------------------------------------------------------------------------
// Block-wide deterministic top-16 (descending, ties -> lower index)
// ---------------------------------------------------------------------------
__global__ __launch_bounds__(256) void topk16_kernel(const float* __restrict__ v,
    int n, int* __restrict__ idx_out, float* __restrict__ val_out, int normalize){
  const int b = blockIdx.x;
  const float* src = v + (size_t)b * (size_t)n;
  const int tid = threadIdx.x;
  float lv[16]; int li[16];
  #pragma unroll
  for (int e = 0; e < 16; ++e){ lv[e] = -3.4e38f; li[e] = 0x7fffffff; }
  for (int i = tid; i < n; i += 256){
    float x = src[i];
    if (x > lv[15] || (x == lv[15] && i < li[15])){
      lv[15] = x; li[15] = i;
      #pragma unroll
      for (int e = 15; e >= 1; --e){
        bool sw = (lv[e] > lv[e-1]) || (lv[e] == lv[e-1] && li[e] < li[e-1]);
        if (sw){
          float tv = lv[e]; lv[e] = lv[e-1]; lv[e-1] = tv;
          int ti = li[e]; li[e] = li[e-1]; li[e-1] = ti;
        }
      }
    }
  }
  __shared__ float rv[256]; __shared__ int ri[256]; __shared__ int rt[256];
  __shared__ int win;
  int ptr = 0;
  for (int p = 0; p < 16; ++p){
    rv[tid] = (ptr < 16) ? lv[ptr] : -3.4e38f;
    ri[tid] = (ptr < 16) ? li[ptr] : 0x7fffffff;
    rt[tid] = tid;
    __syncthreads();
    for (int s = 128; s > 0; s >>= 1){
      if (tid < s){
        bool take = (rv[tid+s] > rv[tid]) || (rv[tid+s] == rv[tid] && ri[tid+s] < ri[tid]);
        if (take){ rv[tid] = rv[tid+s]; ri[tid] = ri[tid+s]; rt[tid] = rt[tid+s]; }
      }
      __syncthreads();
    }
    if (tid == 0){ idx_out[b*16 + p] = ri[0]; val_out[b*16 + p] = rv[0]; win = rt[0]; }
    __syncthreads();
    if (tid == win) ptr++;
    __syncthreads();
  }
  if (normalize && tid == 0){
    float s = 1e-8f;
    for (int p = 0; p < 16; ++p) s += val_out[b*16 + p];
    float inv = 1.f / s;
    for (int p = 0; p < 16; ++p) val_out[b*16 + p] *= inv;
  }
}

__global__ __launch_bounds__(128) void events_kernel(const u16* __restrict__ hN,
    const int* __restrict__ ai, const float* __restrict__ av, u16* __restrict__ E){
  int k = blockIdx.x, b = blockIdx.y, c = threadIdx.x;
  int r = b * 16 + k;
  int t = ai[r];
  E[r * 128 + c] = f2bfu(bfu2f(hN[(b * CH + c) * T_N + t]) * av[r]);
}

// ---------------------------------------------------------------------------
// hard_softmax == argmax in forward; select envelope/transfer rows, decay,
// and run the 128-step first-order recurrence.
// ---------------------------------------------------------------------------
__global__ __launch_bounds__(128) void select_kernel(const float* __restrict__ logitsE,
    const float* __restrict__ logitsT, const float* __restrict__ dvec,
    const float* __restrict__ envelopes, int* __restrict__ tsel,
    float* __restrict__ ef, float* __restrict__ envf){
  int r = threadIdx.x;
  int ie = 0; float bv = logitsE[r * 512];
  for (int i = 1; i < 512; ++i){ float v = logitsE[r*512+i]; if (v > bv){ bv = v; ie = i; } }
  int it = 0; bv = logitsT[r * 512];
  for (int i = 1; i < 512; ++i){ float v = logitsT[r*512+i]; if (v > bv){ bv = v; it = i; } }
  tsel[r] = it;
  float d  = dvec[r];
  float dd = 0.5f + 0.5f * (1.f / (1.f + expf(-d))) * 0.99f;
  float prev = 0.f;
  for (int i = 0; i < 128; ++i){
    float e = (i < 8) ? fabsf(envelopes[ie * 8 + i]) : 0.f;
    ef[r * 128 + i] = e;
    prev = e + dd * prev;
    envf[r * 128 + i] = prev;
  }
}

// mixture = interp(e)*noise + interp(env)*transfer[sel]
__global__ __launch_bounds__(256) void mixture_kernel(const float* __restrict__ ef,
    const float* __restrict__ envf, const int* __restrict__ tsel,
    const float* __restrict__ transfer, float* __restrict__ mixture){
  int r = blockIdx.z * 16 + blockIdx.y;
  int t = blockIdx.x * 256 + threadIdx.x;
  float pos = (t + 0.5f) * (128.f / 32768.f) - 0.5f;
  pos = fminf(fmaxf(pos, 0.f), 127.f);
  int lo = (int)floorf(pos);
  int hi = (lo + 1 < 127) ? lo + 1 : 127;
  float w = pos - (float)lo;
  float envlp = ef[r*128+lo]   * (1.f - w) + ef[r*128+hi]   * w;
  float wdec  = envf[r*128+lo] * (1.f - w) + envf[r*128+hi] * w;
  u32 s = (u32)(r * T_N + t) * 747796405u + 2891336453u;
  u32 h = ((s >> ((s >> 28) + 4u)) ^ s) * 277803737u;
  h = (h >> 22) ^ h;
  float noise = (float)(h >> 8) * (1.f / 8388608.f) - 1.f;   // [-1,1)
  float tv = transfer[tsel[r] * T_N + t];
  mixture[r * T_N + t] = envlp * noise + wdec * tv;
}

// y[b,t] = sum over 16 global-top-k atoms: val * mixture[b, k, t-pos]
__global__ __launch_bounds__(256) void sparse_conv_kernel(const float* __restrict__ mixture,
    const int* __restrict__ nzidx, const float* __restrict__ nzval,
    float* __restrict__ y){
  __shared__ int sk[16]; __shared__ int sp[16]; __shared__ float sv[16];
  int b = blockIdx.y;
  int tid = threadIdx.x;
  if (tid < 16){
    int f = nzidx[b * 16 + tid];
    sk[tid] = f >> 15; sp[tid] = f & (T_N - 1); sv[tid] = nzval[b * 16 + tid];
  }
  __syncthreads();
  int t = blockIdx.x * 256 + tid;
  float a = 0.f;
  #pragma unroll
  for (int i = 0; i < 16; ++i){
    int tt = t - sp[i];
    if (tt >= 0) a += sv[i] * mixture[(b * 16 + sk[i]) * T_N + tt];
  }
  y[b * T_N + t] = a;
}

__global__ __launch_bounds__(256) void head_kernel(const float* __restrict__ ctx,
    const float* __restrict__ room_w, const float* __restrict__ room_b,
    const float* __restrict__ mix_w, const float* __restrict__ mix_b,
    float* __restrict__ rsel, float* __restrict__ mx){
  __shared__ float lg[64]; __shared__ float lm[16];
  int tid = threadIdx.x;
  if (tid < 64){
    int b = tid >> 3, r = tid & 7;
    float a = room_b[r];
    for (int c = 0; c < 128; ++c) a += ctx[b*128 + c] * room_w[r*128 + c];
    lg[tid] = a;
  } else if (tid < 80){
    int q = tid - 64; int b = q >> 1, j = q & 1;
    float a = mix_b[j];
    for (int c = 0; c < 128; ++c) a += ctx[b*128 + c] * mix_w[j*128 + c];
    lm[q] = a;
  }
  __syncthreads();
  if (tid < 8){
    int b = tid;
    float m0 = lg[b*8];
    for (int r = 1; r < 8; ++r) m0 = fmaxf(m0, lg[b*8+r]);
    float s = 0.f, e[8];
    for (int r = 0; r < 8; ++r){ e[r] = expf(lg[b*8+r] - m0); s += e[r]; }
    for (int r = 0; r < 8; ++r) rsel[b*8+r] = e[r] / s;
    float mm = fmaxf(lm[b*2], lm[b*2+1]);
    float e0 = expf(lm[b*2] - mm), e1 = expf(lm[b*2+1] - mm);
    mx[b*2]   = e0 / (e0 + e1);
    mx[b*2+1] = e1 / (e0 + e1);
  }
}

__global__ __launch_bounds__(256) void imp_kernel(const float* __restrict__ rsel,
    const float* __restrict__ rooms, float* __restrict__ imp){
  int b = blockIdx.y;
  int t = blockIdx.x * 256 + threadIdx.x;
  float a = 0.f;
  #pragma unroll
  for (int r = 0; r < 8; ++r) a += rsel[b*8 + r] * rooms[r * T_N + t];
  imp[b * T_N + t] = a;
}

// Causal (FFT-equivalent) reverb conv: wet[t] = sum_{tau<=t} y[tau]*imp[t-tau]
__global__ __launch_bounds__(128) void reverb_kernel(const float* __restrict__ y,
    const float* __restrict__ imp, float* __restrict__ wet){
  __shared__ float ys[128];
  __shared__ float is_[256];
  int b  = blockIdx.y;
  int t0 = blockIdx.x * 128;
  int tid = threadIdx.x;
  float acc = 0.f;
  for (int q0 = 0; q0 <= t0; q0 += 128){
    __syncthreads();
    ys[tid] = y[b * T_N + q0 + tid];
    int base = t0 - q0 - 127;
    int g = base + tid;
    is_[tid] = (g >= 0) ? imp[b * T_N + g] : 0.f;
    if (tid < 127){
      int g2 = base + 128 + tid;
      is_[128 + tid] = (g2 >= 0 && g2 < T_N) ? imp[b * T_N + g2] : 0.f;
    }
    __syncthreads();
    #pragma unroll 4
    for (int i = 0; i < 128; ++i) acc += ys[i] * is_[tid - i + 127];
  }
  wet[b * T_N + t0 + tid] = acc;
}

__global__ __launch_bounds__(256) void final_kernel(const float* __restrict__ y,
    const float* __restrict__ wet, const float* __restrict__ mx,
    float* __restrict__ out){
  int i = blockIdx.x * 256 + threadIdx.x;
  int b = i >> 15;
  out[i] = y[i] * mx[b*2] + wet[i] * mx[b*2 + 1];
}

// ===========================================================================
// Host-side launch
// ===========================================================================
extern "C" void kernel_launch(void* const* d_in, const int* in_sizes, int n_in,
                              void* d_out, int out_size, void* d_ws, size_t ws_size,
                              hipStream_t stream){
  (void)in_sizes; (void)out_size; (void)ws_size;
  const float* x        = (const float*)d_in[0];
  const float* fb_w     = (const float*)d_in[1];
  const float* reduce_w = (const float*)d_in[2];
  const float* reduce_b = (const float*)d_in[3];
  const float* dil_w    = (const float*)d_in[4];
  const float* dil_bv   = (const float*)d_in[5];
  const float* attn_w   = (const float*)d_in[6];
  const float* attn_b   = (const float*)d_in[7];
  const float* atom_w   = (const float*)d_in[8];
  const float* atom_b   = (const float*)d_in[9];
  const float* envelopes= (const float*)d_in[10];
  const float* transfer = (const float*)d_in[11];

  const float *ew1,*eb1,*ew2,*eb2,*ew3,*eb3;
  const float *tw1,*tb1,*tw2,*tb2,*tw3,*tb3;
  const float *dw1,*db1,*dw2,*db2,*dw3,*db3;
  const float *cw1,*cb1,*cw2,*cb2,*cw3,*cb3;
  const float *rooms,*room_w,*room_b,*mix_w,*mix_b;
  if (n_in >= 41){
    ew1=(const float*)d_in[12]; eb1=(const float*)d_in[13];
    ew2=(const float*)d_in[14]; eb2=(const float*)d_in[15];
    ew3=(const float*)d_in[16]; eb3=(const float*)d_in[17];
    tw1=(const float*)d_in[18]; tb1=(const float*)d_in[19];
    tw2=(const float*)d_in[20]; tb2=(const float*)d_in[21];
    tw3=(const float*)d_in[22]; tb3=(const float*)d_in[23];
    dw1=(const float*)d_in[24]; db1=(const float*)d_in[25];
    dw2=(const float*)d_in[26]; db2=(const float*)d_in[27];
    dw3=(const float*)d_in[28]; db3=(const float*)d_in[29];
    cw1=(const float*)d_in[30]; cb1=(const float*)d_in[31];
    cw2=(const float*)d_in[32]; cb2=(const float*)d_in[33];
    cw3=(const float*)d_in[34]; cb3=(const float*)d_in[35];
    rooms=(const float*)d_in[36]; room_w=(const float*)d_in[37];
    room_b=(const float*)d_in[38]; mix_w=(const float*)d_in[39];
    mix_b=(const float*)d_in[40];
  } else {
    // tuples passed as single concatenated leaves (w1,b1,w2,b2,w3,b3)
    const float* p;
    p=(const float*)d_in[12]; ew1=p; eb1=p+16384; ew2=eb1+128; eb2=ew2+16384; ew3=eb2+128; eb3=ew3+65536;
    p=(const float*)d_in[13]; tw1=p; tb1=p+16384; tw2=tb1+128; tb2=tw2+16384; tw3=tb2+128; tb3=tw3+65536;
    p=(const float*)d_in[14]; dw1=p; db1=p+16384; dw2=db1+128; db2=dw2+16384; dw3=db2+128; db3=dw3+128;
    p=(const float*)d_in[15]; cw1=p; cb1=p+16384; cw2=cb1+128; cb2=cw2+16384; cw3=cb2+128; cb3=cw3+16384;
    rooms=(const float*)d_in[16]; room_w=(const float*)d_in[17];
    room_b=(const float*)d_in[18]; mix_w=(const float*)d_in[19];
    mix_b=(const float*)d_in[20];
  }

  char* base = (char*)d_ws; size_t off = 0;
  auto alloc = [&](size_t bytes) -> void* {
    size_t a = (off + 255) & ~(size_t)255;
    off = a + bytes;
    return (void*)(base + a);
  };

  const size_t HN = (size_t)BATCH * CH * T_N;     // 33,554,432
  u16* hA   = (u16*)alloc(HN * 2);
  u16* hB   = (u16*)alloc(HN * 2);
  u16* peT  = (u16*)alloc((size_t)33 * T_N * 2);
  u16* wfb  = (u16*)alloc((size_t)128 * 512 * 2);
  u16* wred = (u16*)alloc((size_t)128 * 192 * 2);
  u16* wconv= (u16*)alloc((size_t)6 * 128 * 384 * 2);
  u16* wE1  = (u16*)alloc(16384 * 2);
  u16* wE2  = (u16*)alloc(16384 * 2);
  u16* wE3  = (u16*)alloc(65536 * 2);
  u16* wT1  = (u16*)alloc(16384 * 2);
  u16* wT2  = (u16*)alloc(16384 * 2);
  u16* wT3  = (u16*)alloc(65536 * 2);
  u16* wD1  = (u16*)alloc(16384 * 2);
  u16* wD2  = (u16*)alloc(16384 * 2);
  u16* wD3  = (u16*)alloc(128 * 2);
  u16* wC1  = (u16*)alloc(16384 * 2);
  u16* wC2  = (u16*)alloc(16384 * 2);
  u16* wC3  = (u16*)alloc(16384 * 2);
  u16* Ev   = (u16*)alloc(16384 * 2);
  u16* L1b  = (u16*)alloc(16384 * 2);
  u16* L2b  = (u16*)alloc(16384 * 2);
  u16* ctxi = (u16*)alloc(2048 * 2);
  float* attn   = (float*)alloc((size_t)8 * T_N * 4);
  float* am     = (float*)alloc((size_t)8 * 16 * T_N * 4);
  float* mixt   = (float*)alloc((size_t)128 * T_N * 4);
  float* logE   = (float*)alloc(128 * 512 * 4);
  float* logT   = (float*)alloc(128 * 512 * 4);
  float* dvec   = (float*)alloc(128 * 4);
  float* rowsum = (float*)alloc(1024 * 4);
  float* rowsq  = (float*)alloc(1024 * 4);
  float* istd   = (float*)alloc(8 * 4);
  float* ctxF   = (float*)alloc(16 * 128 * 4);
  float* ef     = (float*)alloc(128 * 128 * 4);
  float* envf   = (float*)alloc(128 * 128 * 4);
  float* ybuf   = (float*)alloc((size_t)8 * T_N * 4);
  float* wetb   = (float*)alloc((size_t)8 * T_N * 4);
  float* impb   = (float*)alloc((size_t)8 * T_N * 4);
  float* rsel   = (float*)alloc(64 * 4);
  float* mxb    = (float*)alloc(16 * 4);
  int*   ai     = (int*)alloc(128 * 4);
  float* av     = (float*)alloc(128 * 4);
  int*   nzidx  = (int*)alloc(128 * 4);
  float* nzval  = (float*)alloc(128 * 4);
  int*   tsel   = (int*)alloc(128 * 4);

  // ---- weight packing ----
  pack_bf_kernel  <<<(65536 + 255) / 256, 256, 0, stream>>>(fb_w, wfb, 65536);
  pack_pad_kernel <<<(128*192 + 255) / 256, 256, 0, stream>>>(reduce_w, wred, 128, 161, 192);
  pack_conv_kernel<<<(6*128*384 + 255) / 256, 256, 0, stream>>>(dil_w, wconv);
  pack_tr_kernel  <<<64, 256, 0, stream>>>(ew1, wE1, 128, 128);
  pack_tr_kernel  <<<64, 256, 0, stream>>>(ew2, wE2, 128, 128);
  pack_tr_kernel  <<<256, 256, 0, stream>>>(ew3, wE3, 512, 128);
  pack_tr_kernel  <<<64, 256, 0, stream>>>(tw1, wT1, 128, 128);
  pack_tr_kernel  <<<64, 256, 0, stream>>>(tw2, wT2, 128, 128);
  pack_tr_kernel  <<<256, 256, 0, stream>>>(tw3, wT3, 512, 128);
  pack_tr_kernel  <<<64, 256, 0, stream>>>(dw1, wD1, 128, 128);
  pack_tr_kernel  <<<64, 256, 0, stream>>>(dw2, wD2, 128, 128);
  pack_tr_kernel  <<<1, 256, 0, stream>>>(dw3, wD3, 1, 128);
  pack_tr_kernel  <<<64, 256, 0, stream>>>(cw1, wC1, 128, 128);
  pack_tr_kernel  <<<64, 256, 0, stream>>>(cw2, wC2, 128, 128);
  pack_tr_kernel  <<<64, 256, 0, stream>>>(cw3, wC3, 128, 128);
  pe_kernel       <<<128, 256, 0, stream>>>(peT);

  // ---- WMMA backbone ----
  fb_conv_kernel<<<dim3(512, 1, 8), 256, 0, stream>>>(x, wfb, hB);
  reduce_kernel <<<dim3(512, 1, 8), 256, 0, stream>>>(peT, hB, wred, reduce_b, hA);
  const int dils[6] = {1, 3, 9, 27, 81, 1};
  u16* cin = hA; u16* cout = hB;
  for (int l = 0; l < 6; ++l){
    dconv_kernel<<<dim3(256, 1, 8), 256, 0, stream>>>(
        cin, wconv + (size_t)l * 128 * 384, dil_bv + l * 128, cout, dils[l]);
    u16* tmp = cin; cin = cout; cout = tmp;
  }
  // result in cin (== hA)

  // ---- ExampleNorm + context input ----
  rowstat_kernel  <<<1024, 256, 0, stream>>>(cin, rowsum, rowsq);
  batchstat_kernel<<<16, 128, 0, stream>>>(rowsum, rowsq, istd, ctxi);
  scale_kernel    <<<65536, 256, 0, stream>>>(cin, cout, istd);   // hN in cout

  // ---- projections, top-k, events ----
  attn_am_kernel<<<dim3(256, 8), 128, 0, stream>>>(cout, attn_w, attn_b,
                                                   atom_w, atom_b, attn, am);
  topk16_kernel <<<8, 256, 0, stream>>>(attn, T_N, ai, av, 1);
  topk16_kernel <<<8, 256, 0, stream>>>(am, 16 * T_N, nzidx, nzval, 0);
  events_kernel <<<dim3(16, 8), 128, 0, stream>>>(cout, ai, av, Ev);

  // ---- MLP stacks (WMMA) ----
  // ctx (M=16, rows 8..15 zero)
  gemm_stack_kernel<<<2, 256, 0, stream>>>(ctxi, wC1, cb1, 16, 128, 128, 1, nullptr, L1b);
  gemm_stack_kernel<<<2, 256, 0, stream>>>(L1b,  wC2, cb2, 16, 128, 128, 1, nullptr, L2b);
  gemm_stack_kernel<<<2, 256, 0, stream>>>(L2b,  wC3, cb3, 16, 128, 128, 0, ctxF, nullptr);
  // env
  gemm_stack_kernel<<<2, 256, 0, stream>>>(Ev,  wE1, eb1, 128, 128, 128, 1, nullptr, L1b);
  gemm_stack_kernel<<<2, 256, 0, stream>>>(L1b, wE2, eb2, 128, 128, 128, 1, nullptr, L2b);
  gemm_stack_kernel<<<8, 256, 0, stream>>>(L2b, wE3, eb3, 128, 512, 128, 0, logE, nullptr);
  // trans
  gemm_stack_kernel<<<2, 256, 0, stream>>>(Ev,  wT1, tb1, 128, 128, 128, 1, nullptr, L1b);
  gemm_stack_kernel<<<2, 256, 0, stream>>>(L1b, wT2, tb2, 128, 128, 128, 1, nullptr, L2b);
  gemm_stack_kernel<<<8, 256, 0, stream>>>(L2b, wT3, tb3, 128, 512, 128, 0, logT, nullptr);
  // decay
  gemm_stack_kernel<<<2, 256, 0, stream>>>(Ev,  wD1, db1, 128, 128, 128, 1, nullptr, L1b);
  gemm_stack_kernel<<<2, 256, 0, stream>>>(L1b, wD2, db2, 128, 128, 128, 1, nullptr, L2b);
  gemm_stack_kernel<<<1, 256, 0, stream>>>(L2b, wD3, db3, 128, 1, 128, 0, dvec, nullptr);

  // ---- synthesis ----
  select_kernel     <<<1, 128, 0, stream>>>(logE, logT, dvec, envelopes, tsel, ef, envf);
  mixture_kernel    <<<dim3(128, 16, 8), 256, 0, stream>>>(ef, envf, tsel, transfer, mixt);
  sparse_conv_kernel<<<dim3(128, 8), 256, 0, stream>>>(mixt, nzidx, nzval, ybuf);
  head_kernel       <<<1, 256, 0, stream>>>(ctxF, room_w, room_b, mix_w, mix_b, rsel, mxb);
  imp_kernel        <<<dim3(128, 8), 256, 0, stream>>>(rsel, rooms, impb);
  reverb_kernel     <<<dim3(256, 8), 128, 0, stream>>>(ybuf, impb, wetb);
  final_kernel      <<<1024, 256, 0, stream>>>(ybuf, wetb, mxb, (float*)d_out);
}
